// InternalDecoder55_84447646974515
// MI455X (gfx1250) — matrix operand order repaired and checked by reading, not compiled, not run on hardware
//
#include <hip/hip_runtime.h>
#include <hip/hip_bf16.h>

#define N_NODES   20000
#define FEAT      256
#define N_RBF     16
#define E_UND     160000
#define E_DIR     320000
#define NUM_CONV  3
#define CUTOFF_F  10.0f
#define PI_F      3.14159265358979323846f

typedef __attribute__((ext_vector_type(16))) __bf16 v16bf;
typedef __attribute__((ext_vector_type(8)))  float  v8f;

__device__ __forceinline__ float silu_f(float x) { return x / (1.0f + __expf(-x)); }

// ---------------------------------------------------------------------------
// Transpose 256x256 f32 weight [k][f] -> bf16 [f][k]  (one launch per matrix)
// ---------------------------------------------------------------------------
__global__ void k_transpose_bf16(const float* __restrict__ W, __bf16* __restrict__ Wt) {
    int t = blockIdx.x * blockDim.x + threadIdx.x;   // 65536 elements
    if (t >= FEAT * FEAT) return;
    int f = t >> 8;
    int k = t & 255;
    Wt[t] = (__bf16)W[k * FEAT + f];
}

// ---------------------------------------------------------------------------
// Per-node embeddings / small angle MLP
// ---------------------------------------------------------------------------
__global__ void k_node_prep(const int* __restrict__ cg_z, const float* __restrict__ Sin,
                            const float* __restrict__ bb_dist_emb, const float* __restrict__ sc_dist_emb,
                            const float* __restrict__ sc_angle_emb,
                            const float* __restrict__ bba_w1, const float* __restrict__ bba_b1,
                            const float* __restrict__ bba_w2, const float* __restrict__ bba_b2,
                            float* __restrict__ bbd, float* __restrict__ bba,
                            float* __restrict__ scd, float* __restrict__ sca) {
    int n = blockIdx.x * blockDim.x + threadIdx.x;
    if (n >= N_NODES) return;
    int z = cg_z[n];
    #pragma unroll
    for (int r = 0; r < 3; ++r)  bbd[n * 3 + r]  = fabsf(bb_dist_emb[z * 3 + r]);
    #pragma unroll
    for (int r = 0; r < 10; ++r) scd[n * 10 + r] = fabsf(sc_dist_emb[z * 10 + r]);
    #pragma unroll
    for (int r = 0; r < 10; ++r) sca[n * 10 + r] = sc_angle_emb[z * 10 + r];

    float ain[4] = { Sin[n * 253 + 0], Sin[n * 253 + 1], Sin[n * 253 + 2], (float)z };
    float h[3];
    #pragma unroll
    for (int o = 0; o < 3; ++o) {
        float s = bba_b1[o];
        #pragma unroll
        for (int i = 0; i < 4; ++i) s += silu_f(ain[i]) * bba_w1[i * 3 + o];
        h[o] = silu_f(s);
    }
    #pragma unroll
    for (int o = 0; o < 3; ++o) {
        float s = bba_b2[o];
        #pragma unroll
        for (int j = 0; j < 3; ++j) s += h[j] * bba_w2[j * 3 + o];
        bba[n * 3 + o] = s;
    }
}

// S0[n][k] = k<250 ? Sin[n][3+k] : res_embed[z[n]][k-250]
__global__ void k_build_S0(const int* __restrict__ cg_z, const float* __restrict__ Sin,
                           const float* __restrict__ res_embed, float* __restrict__ S0) {
    int t = blockIdx.x * blockDim.x + threadIdx.x;
    if (t >= N_NODES * FEAT) return;
    int n = t >> 8;
    int k = t & 255;
    S0[t] = (k < 250) ? Sin[n * 253 + 3 + k] : res_embed[cg_z[n] * 6 + (k - 250)];
}

// ---------------------------------------------------------------------------
// Edge preprocessing: directed edges, rbf(16), envelope
// ---------------------------------------------------------------------------
__global__ void k_edge_prep(const int* __restrict__ nbrs, const float* __restrict__ xyz,
                            int* __restrict__ iidx, int* __restrict__ jidx,
                            float* __restrict__ rbf, float* __restrict__ env) {
    int e = blockIdx.x * blockDim.x + threadIdx.x;
    if (e >= E_DIR) return;
    int i, j;
    if (e < E_UND) { i = nbrs[2 * e];             j = nbrs[2 * e + 1]; }
    else           { int u = e - E_UND; i = nbrs[2 * u + 1]; j = nbrs[2 * u]; }
    iidx[e] = i; jidx[e] = j;
    float dx = xyz[3 * j + 0] - xyz[3 * i + 0];
    float dy = xyz[3 * j + 1] - xyz[3 * i + 1];
    float dz = xyz[3 * j + 2] - xyz[3 * i + 2];
    float d = sqrtf(dx * dx + dy * dy + dz * dz);
    float dinv = (d > 1e-12f) ? (1.0f / d) : 0.0f;
    #pragma unroll
    for (int m = 0; m < N_RBF; ++m)
        rbf[(size_t)e * N_RBF + m] = sinf((float)(m + 1) * (PI_F / CUTOFF_F) * d) * dinv;
    env[e] = (d < CUTOFF_F) ? 0.5f * (cosf(PI_F * d / CUTOFF_F) + 1.0f) : 0.0f;
}

// ---------------------------------------------------------------------------
// f32 -> bf16 activation convert (optional silu)
// ---------------------------------------------------------------------------
__global__ void k_act_bf16(const float* __restrict__ in, __bf16* __restrict__ out, int n, int do_silu) {
    int t = blockIdx.x * blockDim.x + threadIdx.x;
    if (t >= n) return;
    float x = in[t];
    out[t] = (__bf16)(do_silu ? silu_f(x) : x);
}

__global__ void k_zero_f32(float* __restrict__ p, int n) {
    int t = blockIdx.x * blockDim.x + threadIdx.x;
    if (t < n) p[t] = 0.0f;
}

// ---------------------------------------------------------------------------
// WMMA GEMM: C(20000x256) = A(bf16, 20000x256) @ W(256x256) + bias, epilogue:
//   mode 0: outF = val            (f32)
//   mode 1: outB = bf16(silu(val))
//   mode 2: outF = res + val      (residual, f32)
// Wt is bf16, transposed [out_f][in_k].
// Each wave computes a 2x2 group of 16x16 tiles (32x32 output): A fragments
// are reused across the two N-tiles and B fragments across the two M-tiles,
// halving bytes/FLOP. 1250 M-tiles -> 625 M-groups; 16 N-tiles -> 8 N-groups.
// ---------------------------------------------------------------------------
#define MT_GROUPS (N_NODES / 32)    // 625
#define NT_GROUPS (FEAT / 32)       // 8

__global__ void k_gemm256(const __bf16* __restrict__ A, const __bf16* __restrict__ Wt,
                          const float* __restrict__ bias, const float* __restrict__ res,
                          float* __restrict__ outF, __bf16* __restrict__ outB, int mode) {
    const int wpb   = blockDim.x >> 5;
    const int group = blockIdx.x * wpb + (threadIdx.x >> 5);   // 5000 groups total
    if (group >= MT_GROUPS * NT_GROUPS) return;
    const int mg = group >> 3;          // 0..624
    const int ng = group & 7;           // 0..7
    const int lane = threadIdx.x & 31;
    const int half = lane >> 4;         // 0 or 1
    const int lrow = lane & 15;

    v8f acc00 = {0.f,0.f,0.f,0.f,0.f,0.f,0.f,0.f};
    v8f acc01 = acc00, acc10 = acc00, acc11 = acc00;

    const __bf16* arow0 = A  + (size_t)(mg * 32 + lrow) * FEAT;        // M-tile 0
    const __bf16* arow1 = arow0 + (size_t)16 * FEAT;                   // M-tile 1
    const __bf16* brow0 = Wt + (size_t)(ng * 32 + lrow) * FEAT;        // N-tile 0
    const __bf16* brow1 = brow0 + (size_t)16 * FEAT;                   // N-tile 1

    #pragma unroll
    for (int k0 = 0; k0 < FEAT; k0 += 32) {
        union { v16bf v; uint4 u[2]; } a0, a1, b0, b1;
        // A frag: K groups {base..base+7, base+16..base+23}, base = k0 + half*8
        const int aoff = k0 + half * 8;
        a0.u[0] = *(const uint4*)(arow0 + aoff);
        a0.u[1] = *(const uint4*)(arow0 + aoff + 16);
        a1.u[0] = *(const uint4*)(arow1 + aoff);
        a1.u[1] = *(const uint4*)(arow1 + aoff + 16);
        // B frag: K groups {base..base+7, base+16..base+23}, base = k0 + half*8
        const int boff = k0 + half * 8;
        b0.u[0] = *(const uint4*)(brow0 + boff);
        b0.u[1] = *(const uint4*)(brow0 + boff + 16);
        b1.u[0] = *(const uint4*)(brow1 + boff);
        b1.u[1] = *(const uint4*)(brow1 + boff + 16);
        acc00 = __builtin_amdgcn_wmma_f32_16x16x32_bf16(false, a0.v, false, b0.v, (short)0, acc00, false, false);
        acc01 = __builtin_amdgcn_wmma_f32_16x16x32_bf16(false, a0.v, false, b1.v, (short)0, acc01, false, false);
        acc10 = __builtin_amdgcn_wmma_f32_16x16x32_bf16(false, a1.v, false, b0.v, (short)0, acc10, false, false);
        acc11 = __builtin_amdgcn_wmma_f32_16x16x32_bf16(false, a1.v, false, b1.v, (short)0, acc11, false, false);
    }

    v8f accs[2][2] = { { acc00, acc01 }, { acc10, acc11 } };
    #pragma unroll
    for (int i = 0; i < 2; ++i) {
        #pragma unroll
        for (int jt = 0; jt < 2; ++jt) {
            const int col  = ng * 32 + jt * 16 + lrow;
            const float bcol = bias[col];
            #pragma unroll
            for (int r = 0; r < 8; ++r) {
                int row = mg * 32 + i * 16 + r + half * 8;
                size_t off = (size_t)row * FEAT + col;
                float val = accs[i][jt][r] + bcol;
                if (mode == 0)      outF[off] = val;
                else if (mode == 1) outB[off] = (__bf16)silu_f(val);
                else                outF[off] = res[off] + val;
            }
        }
    }
}

// ---------------------------------------------------------------------------
// Edge message + segment-sum scatter: v[i] += phi[j] * ((rbf@Wr + br) * env)
// One wave per edge iteration; rbf_w (16x256) cached in LDS.
// ---------------------------------------------------------------------------
__global__ void k_edge_msg(const int* __restrict__ iidx, const int* __restrict__ jidx,
                           const float* __restrict__ rbf, const float* __restrict__ env,
                           const float* __restrict__ rbf_w, const float* __restrict__ rbf_b,
                           const float* __restrict__ phi, float* __restrict__ v) {
    __shared__ float sW[N_RBF * FEAT];
    __shared__ float sB[FEAT];
    for (int t = threadIdx.x; t < N_RBF * FEAT; t += blockDim.x) sW[t] = rbf_w[t];
    for (int t = threadIdx.x; t < FEAT; t += blockDim.x)         sB[t] = rbf_b[t];
    __syncthreads();

    const int lane   = threadIdx.x & 31;
    const int wpb    = blockDim.x >> 5;
    const int nwaves = gridDim.x * wpb;
    for (int e = blockIdx.x * wpb + (threadIdx.x >> 5); e < E_DIR; e += nwaves) {
        const int i = iidx[e], j = jidx[e];
        const float envv = env[e];
        float rb[N_RBF];
        const float4* rp = (const float4*)(rbf + (size_t)e * N_RBF);
        #pragma unroll
        for (int q = 0; q < 4; ++q) {
            float4 r4 = rp[q];
            rb[4 * q + 0] = r4.x; rb[4 * q + 1] = r4.y;
            rb[4 * q + 2] = r4.z; rb[4 * q + 3] = r4.w;
        }
        const float* ph = phi + (size_t)j * FEAT;
        float*       vi = v   + (size_t)i * FEAT;
        #pragma unroll
        for (int t = 0; t < FEAT / 32; ++t) {
            int f = lane + t * 32;
            float w = 0.0f;
            #pragma unroll
            for (int m = 0; m < N_RBF; ++m) w += rb[m] * sW[m * FEAT + f];
            float ws = (w + sB[f]) * envv;
            unsafeAtomicAdd(vi + f, ph[f] * ws);
        }
    }
}

// ---------------------------------------------------------------------------
// Heads: bb_torsion (256->3->3), sc_torsion (256->10->10), assemble (N,13,3)
// One wave per node.
// ---------------------------------------------------------------------------
__global__ void k_head(const float* __restrict__ S,
                       const float* __restrict__ bbt_w1, const float* __restrict__ bbt_b1,
                       const float* __restrict__ bbt_w2, const float* __restrict__ bbt_b2,
                       const float* __restrict__ ft_w1,  const float* __restrict__ ft_b1,
                       const float* __restrict__ ft_w2,  const float* __restrict__ ft_b2,
                       const float* __restrict__ bbd, const float* __restrict__ bba,
                       const float* __restrict__ scd, const float* __restrict__ sca,
                       float* __restrict__ out) {
    const int wpb = blockDim.x >> 5;
    const int n = blockIdx.x * wpb + (threadIdx.x >> 5);
    if (n >= N_NODES) return;
    const int lane = threadIdx.x & 31;

    float ab[3]  = {0.f, 0.f, 0.f};
    float af[10] = {0.f, 0.f, 0.f, 0.f, 0.f, 0.f, 0.f, 0.f, 0.f, 0.f};
    for (int k = lane; k < FEAT; k += 32) {
        float x = silu_f(S[(size_t)n * FEAT + k]);
        #pragma unroll
        for (int o = 0; o < 3; ++o)  ab[o] += x * bbt_w1[k * 3 + o];
        #pragma unroll
        for (int o = 0; o < 10; ++o) af[o] += x * ft_w1[k * 10 + o];
    }
    #pragma unroll
    for (int s = 16; s > 0; s >>= 1) {
        #pragma unroll
        for (int o = 0; o < 3; ++o)  ab[o] += __shfl_xor(ab[o], s, 32);
        #pragma unroll
        for (int o = 0; o < 10; ++o) af[o] += __shfl_xor(af[o], s, 32);
    }
    if (lane == 0) {
        float hb[3], bt[3], hf[10], st[10];
        #pragma unroll
        for (int o = 0; o < 3; ++o) hb[o] = silu_f(ab[o] + bbt_b1[o]);
        #pragma unroll
        for (int o = 0; o < 3; ++o) {
            float s = bbt_b2[o];
            #pragma unroll
            for (int j = 0; j < 3; ++j) s += hb[j] * bbt_w2[j * 3 + o];
            bt[o] = s;
        }
        #pragma unroll
        for (int o = 0; o < 10; ++o) hf[o] = silu_f(af[o] + ft_b1[o]);
        #pragma unroll
        for (int o = 0; o < 10; ++o) {
            float s = ft_b2[o];
            #pragma unroll
            for (int j = 0; j < 10; ++j) s += hf[j] * ft_w2[j * 10 + o];
            st[o] = s;
        }
        float* po = out + (size_t)n * 39;
        #pragma unroll
        for (int r = 0; r < 3; ++r) {
            po[r * 3 + 0] = bbd[n * 3 + r];
            po[r * 3 + 1] = bba[n * 3 + r];
            po[r * 3 + 2] = bt[r];
        }
        #pragma unroll
        for (int r = 0; r < 10; ++r) {
            po[(3 + r) * 3 + 0] = scd[n * 10 + r];
            po[(3 + r) * 3 + 1] = sca[n * 10 + r];
            po[(3 + r) * 3 + 2] = st[r];
        }
    }
}

// ---------------------------------------------------------------------------
// Host side
// ---------------------------------------------------------------------------
static inline size_t alignup(size_t x) { return (x + 255) & ~(size_t)255; }

extern "C" void kernel_launch(void* const* d_in, const int* in_sizes, int n_in,
                              void* d_out, int out_size, void* d_ws, size_t ws_size,
                              hipStream_t stream) {
    (void)in_sizes; (void)n_in; (void)out_size; (void)ws_size;
    const int*   cg_z        = (const int*)  d_in[0];
    const float* cg_xyz      = (const float*)d_in[1];
    const int*   nbrs        = (const int*)  d_in[2];
    // d_in[3] = mapping (unused by reference)
    const float* S_in        = (const float*)d_in[4];
    const float* res_embed   = (const float*)d_in[5];
    const float* msg_w1      = (const float*)d_in[6];
    const float* msg_b1      = (const float*)d_in[7];
    const float* msg_w2      = (const float*)d_in[8];
    const float* msg_b2      = (const float*)d_in[9];
    const float* rbf_w       = (const float*)d_in[10];
    const float* rbf_b       = (const float*)d_in[11];
    const float* dense_w1    = (const float*)d_in[12];
    const float* dense_b1    = (const float*)d_in[13];
    const float* dense_w2    = (const float*)d_in[14];
    const float* dense_b2    = (const float*)d_in[15];
    const float* tor_w1      = (const float*)d_in[16];
    const float* tor_b1      = (const float*)d_in[17];
    const float* tor_w2      = (const float*)d_in[18];
    const float* tor_b2      = (const float*)d_in[19];
    const float* bb_dist_emb = (const float*)d_in[20];
    const float* sc_dist_emb = (const float*)d_in[21];
    const float* bba_w1      = (const float*)d_in[22];
    const float* bba_b1      = (const float*)d_in[23];
    const float* bba_w2      = (const float*)d_in[24];
    const float* bba_b2      = (const float*)d_in[25];
    const float* sc_angle_emb= (const float*)d_in[26];
    const float* bbt_w1      = (const float*)d_in[27];
    const float* bbt_b1      = (const float*)d_in[28];
    const float* bbt_w2      = (const float*)d_in[29];
    const float* bbt_b2      = (const float*)d_in[30];
    const float* ft_w1       = (const float*)d_in[31];
    const float* ft_b1       = (const float*)d_in[32];
    const float* ft_w2       = (const float*)d_in[33];
    const float* ft_b2       = (const float*)d_in[34];

    // ---- workspace carve ----
    char* p = (char*)d_ws;
    auto take = [&](size_t bytes) { char* r = p; p += alignup(bytes); return r; };
    const size_t NF = (size_t)N_NODES * FEAT;
    float*  S    = (float*) take(NF * 4);
    float*  phi  = (float*) take(NF * 4);
    float*  vbuf = (float*) take(NF * 4);
    __bf16* Abf  = (__bf16*)take(NF * 2);
    __bf16* Hbf  = (__bf16*)take(NF * 2);
    float*  rbf  = (float*) take((size_t)E_DIR * N_RBF * 4);
    float*  env  = (float*) take((size_t)E_DIR * 4);
    int*    iidx = (int*)   take((size_t)E_DIR * 4);
    int*    jidx = (int*)   take((size_t)E_DIR * 4);
    __bf16* Wt   = (__bf16*)take((size_t)18 * FEAT * FEAT * 2);
    float*  bbd  = (float*) take((size_t)N_NODES * 3 * 4);
    float*  bba  = (float*) take((size_t)N_NODES * 3 * 4);
    float*  scd  = (float*) take((size_t)N_NODES * 10 * 4);
    float*  sca  = (float*) take((size_t)N_NODES * 10 * 4);

    const int WSLICE = FEAT * FEAT;                 // 65536
    auto wt_slot = [&](int idx) { return Wt + (size_t)idx * WSLICE; };

    // ---- weight transposes (f32 [k][f] -> bf16 [f][k]) ----
    for (int c = 0; c < NUM_CONV; ++c) {
        k_transpose_bf16<<<WSLICE / 256, 256, 0, stream>>>(msg_w1   + c * WSLICE, wt_slot(c * 6 + 0));
        k_transpose_bf16<<<WSLICE / 256, 256, 0, stream>>>(msg_w2   + c * WSLICE, wt_slot(c * 6 + 1));
        k_transpose_bf16<<<WSLICE / 256, 256, 0, stream>>>(dense_w1 + c * WSLICE, wt_slot(c * 6 + 2));
        k_transpose_bf16<<<WSLICE / 256, 256, 0, stream>>>(dense_w2 + c * WSLICE, wt_slot(c * 6 + 3));
        k_transpose_bf16<<<WSLICE / 256, 256, 0, stream>>>(tor_w1   + c * WSLICE, wt_slot(c * 6 + 4));
        k_transpose_bf16<<<WSLICE / 256, 256, 0, stream>>>(tor_w2   + c * WSLICE, wt_slot(c * 6 + 5));
    }

    // ---- node / edge prep ----
    k_node_prep<<<(N_NODES + 255) / 256, 256, 0, stream>>>(
        cg_z, S_in, bb_dist_emb, sc_dist_emb, sc_angle_emb,
        bba_w1, bba_b1, bba_w2, bba_b2, bbd, bba, scd, sca);
    k_build_S0<<<(int)(NF / 256), 256, 0, stream>>>(cg_z, S_in, res_embed, S);
    k_edge_prep<<<(E_DIR + 255) / 256, 256, 0, stream>>>(nbrs, cg_xyz, iidx, jidx, rbf, env);

    const int GEMM_BLOCKS = (MT_GROUPS * NT_GROUPS) / 8;        // 625 blocks, 8 waves each
    const int NF_BLOCKS   = (int)(NF / 256);

    // ---- conv layers ----
    for (int c = 0; c < NUM_CONV; ++c) {
        // phi = silu(S @ msg_w1 + b1) @ msg_w2 + b2
        k_act_bf16<<<NF_BLOCKS, 256, 0, stream>>>(S, Abf, (int)NF, 0);
        k_gemm256<<<GEMM_BLOCKS, 256, 0, stream>>>(Abf, wt_slot(c * 6 + 0), msg_b1 + c * FEAT,
                                                   nullptr, nullptr, Hbf, 1);
        k_gemm256<<<GEMM_BLOCKS, 256, 0, stream>>>(Hbf, wt_slot(c * 6 + 1), msg_b2 + c * FEAT,
                                                   nullptr, phi, nullptr, 0);
        // v = segment_sum(phi[j] * w_s)
        k_zero_f32<<<NF_BLOCKS, 256, 0, stream>>>(vbuf, (int)NF);
        k_edge_msg<<<2048, 256, 0, stream>>>(iidx, jidx, rbf, env,
                                             rbf_w + c * N_RBF * FEAT, rbf_b + c * FEAT,
                                             phi, vbuf);
        // S += silu(silu(v) @ dense_w1 + b1) @ dense_w2 + b2
        k_act_bf16<<<NF_BLOCKS, 256, 0, stream>>>(vbuf, Abf, (int)NF, 1);
        k_gemm256<<<GEMM_BLOCKS, 256, 0, stream>>>(Abf, wt_slot(c * 6 + 2), dense_b1 + c * FEAT,
                                                   nullptr, nullptr, Hbf, 1);
        k_gemm256<<<GEMM_BLOCKS, 256, 0, stream>>>(Hbf, wt_slot(c * 6 + 3), dense_b2 + c * FEAT,
                                                   S, S, nullptr, 2);
    }

    // ---- torsion residual layers ----
    for (int c = 0; c < NUM_CONV; ++c) {
        k_act_bf16<<<NF_BLOCKS, 256, 0, stream>>>(S, Abf, (int)NF, 1);
        k_gemm256<<<GEMM_BLOCKS, 256, 0, stream>>>(Abf, wt_slot(c * 6 + 4), tor_b1 + c * FEAT,
                                                   nullptr, nullptr, Hbf, 1);
        k_gemm256<<<GEMM_BLOCKS, 256, 0, stream>>>(Hbf, wt_slot(c * 6 + 5), tor_b2 + c * FEAT,
                                                   S, S, nullptr, 2);
    }

    // ---- heads + output assembly ----
    k_head<<<(N_NODES + 7) / 8, 256, 0, stream>>>(
        S, bbt_w1, bbt_b1, bbt_w2, bbt_b2, ft_w1, ft_b1, ft_w2, ft_b2,
        bbd, bba, scd, sca, (float*)d_out);
}
